// BiRecurrentConvCRF4NestedNER_395136991975
// MI455X (gfx1250) — compile-verified, matrix operand order checked by reading, not run on hardware
//
#include <hip/hip_runtime.h>
#include <hip/hip_bf16.h>
#include <math.h>

// ---------------------------------------------------------------------------
// BiRecurrentConvCRF4NestedNER forward loss on gfx1250 (CDNA5, wave32, WMMA)
//
// Shapes: B=32, L=128, C=20, TOKEN_EMBED=300, CHAR_EMBED=50, NUM_FILTERS=200,
//         KERNEL=3, HID=256 (4H=1024), D0=500 (pad->512), D1=512, LABELS=8, NS=6
// ---------------------------------------------------------------------------

typedef __attribute__((ext_vector_type(16))) _Float16 v16h;
typedef __attribute__((ext_vector_type(8)))  float    v8f;

#define BB   32
#define LL   128
#define HID_ 256
#define GATES 1024   // 4*HID
#define KPAD 512     // padded input feature dim for both layers

// ---------------------------------------------------------------------------
// Zero the scalar output accumulator.
__global__ void k_zero(float* p) {
  if (threadIdx.x == 0 && blockIdx.x == 0) p[0] = 0.0f;
}

// ---------------------------------------------------------------------------
// Convert an f32 (rows x srcK) matrix to f16 (rows x dstK), zero-padding K.
__global__ void k_cvt_pad(const float* __restrict__ src, _Float16* __restrict__ dst,
                          int srcK, int dstK) {
  const int r = blockIdx.x;
  for (int k = threadIdx.x; k < dstK; k += blockDim.x) {
    float v = (k < srcK) ? src[(size_t)r * srcK + k] : 0.0f;
    dst[(size_t)r * dstK + k] = (_Float16)v;
  }
}

// ---------------------------------------------------------------------------
// Word embedding gather: inp_f16[token, 0:300] = emb[iv] + (oo!=0)*ooev[oo].
// Also zero the K-pad columns 500..511. One block per token, 320 threads.
__global__ void k_embed(const int* __restrict__ iv, const int* __restrict__ oo,
                        const float* __restrict__ wtab, const float* __restrict__ otab,
                        _Float16* __restrict__ inp) {
  const int tok = blockIdx.x;
  const int tid = threadIdx.x;
  const int wi = iv[tok];
  const int oi = oo[tok];
  if (tid < 300) {
    float v = wtab[(size_t)wi * 300 + tid];
    if (oi != 0) v += otab[(size_t)oi * 300 + tid];
    inp[(size_t)tok * KPAD + tid] = (_Float16)v;
  } else if (tid < 312) {
    inp[(size_t)tok * KPAD + 200 + tid] = (_Float16)0.0f;  // cols 500..511
  }
}

// ---------------------------------------------------------------------------
// Char CNN per token: masked char embeddings -> conv1d(pad 2,2, K=3, 50->200)
// -> max over 22 positions -> sigmoid -> inp_f16[token, 300:500].
// One block (256 thr) per token; char tile staged in LDS.
__global__ void k_charconv(const int* __restrict__ chars,
                           const float* __restrict__ ctab,
                           const float* __restrict__ cw,   // (200,50,3)
                           const float* __restrict__ cb,   // (200)
                           _Float16* __restrict__ inp) {
  __shared__ float ch[20 * 50];
  const int tok = blockIdx.x;
  for (int i = threadIdx.x; i < 20 * 50; i += blockDim.x) {
    const int ci = i / 50, d = i % 50;
    const int idx = chars[(size_t)tok * 20 + ci];
    ch[i] = (idx != 0) ? ctab[(size_t)idx * 50 + d] : 0.0f;
  }
  __syncthreads();
  const int f = threadIdx.x;
  if (f < 200) {
    const float bias = cb[f];
    float best = -1e30f;
    for (int o = 0; o < 22; ++o) {            // output positions
      float s = bias;
      #pragma unroll
      for (int k = 0; k < 3; ++k) {
        const int p = o - 2 + k;              // zero-padding 2 each side
        if (p >= 0 && p < 20) {
          const float* row = &ch[p * 50];
          const float* w = &cw[(size_t)f * 150 + k];  // stride 3 over d
          #pragma unroll
          for (int d = 0; d < 50; ++d) s += row[d] * w[d * 3];
        }
      }
      best = fmaxf(best, s);
    }
    inp[(size_t)tok * KPAD + 300 + f] = (_Float16)(1.0f / (1.0f + expf(-best)));
  }
}

// ---------------------------------------------------------------------------
// WMMA GEMM: C(MxN,f32) = A(MxK,f16 row-major) * W(NxK,f16 row-major)^T + bias[N]
// Block = 256 threads = 8 waves; block tile 32(M) x 256(N); each wave 16x64.
// Fragment layout used here (contiguous K order; A and B use the same one):
//   A 16x32 f16: lane L<16 holds row (m0+L), K=k0..k0+15; lanes 16..31 same rows, K+16.
//   B 32x16 f16 from row-major W(N,K): symmetric (lane&15 -> N column == W row).
//   C/D: VGPR r -> M = r + 8*(lane>>4); N = lane&15.
__global__ void __launch_bounds__(256) k_wmma_gemm(
    const _Float16* __restrict__ A, const _Float16* __restrict__ W,
    const float* __restrict__ bias, float* __restrict__ C,
    int M, int N, int K) {
  const int lane = threadIdx.x & 31;
  const int wv   = threadIdx.x >> 5;                 // 0..7
  const int lo = lane & 15, hi = lane >> 4;
  const int mrow = blockIdx.x * 32 + (wv & 1) * 16;
  const int n0   = blockIdx.y * 256 + (wv >> 1) * 64;

  v8f acc[4] = {};
  const _Float16* aRow = A + (size_t)(mrow + lo) * K + hi * 16;
  for (int kk = 0; kk < K; kk += 32) {
    __builtin_prefetch((const void*)(aRow + kk + 64), 0, 1);  // global_prefetch_b8
    const v16h a = *(const v16h*)(aRow + kk);
    #pragma unroll
    for (int j = 0; j < 4; ++j) {
      const v16h b = *(const v16h*)(W + (size_t)(n0 + j * 16 + lo) * K + kk + hi * 16);
      acc[j] = __builtin_amdgcn_wmma_f32_16x16x32_f16(false, a, false, b,
                                                      (short)0, acc[j], false, false);
    }
  }
  #pragma unroll
  for (int j = 0; j < 4; ++j) {
    const int col = n0 + j * 16 + lo;
    const float bv = bias[col];
    #pragma unroll
    for (int r = 0; r < 8; ++r)
      C[(size_t)(mrow + r + hi * 8) * N + col] = acc[j][r] + bv;
  }
}

// ---------------------------------------------------------------------------
// Persistent LSTM recurrence, one workgroup per direction (grid=2).
// 512 threads = 16 waves. State h/c (f32), h_f16 and gates live in LDS
// (208 KB of the 320 KB WGP LDS — CDNA5 only). W_hh streams from L2 each step.
// Per step: gates(32x1024) = xs[:,t] + h(32x256) @ Whh^T via WMMA
//           (wave w owns both 16-row M tiles x 4 N tiles at [w*64, w*64+64)).
__global__ void __launch_bounds__(512) k_lstm(
    const float* __restrict__ xs,      // (B*L, 2048): col = dir*1024 + gate
    const _Float16* __restrict__ Whh,  // (2048, 256):  row = dir*1024 + gate
    const float* __restrict__ mask,    // (B, L)
    float* __restrict__ outF,          // (B*L, 512): [dir*256 + j]
    _Float16* __restrict__ outH) {     // (B*L, 512)
  const int dir = blockIdx.x;
  extern __shared__ char smem[];
  float*    h  = (float*)smem;                       // 32*256 f32
  float*    c  = h + BB * HID_;                      // 32*256 f32
  float*    gt = c + BB * HID_;                      // 32*1024 f32
  _Float16* hh = (_Float16*)(gt + BB * GATES);       // 32*256 f16

  const int tid = threadIdx.x;
  const int lane = tid & 31, wv = tid >> 5;          // 16 waves
  const int lo = lane & 15, hi = lane >> 4;
  const int nbase = wv * 64;

  for (int i = tid; i < BB * HID_; i += 512) {
    h[i] = 0.0f; c[i] = 0.0f; hh[i] = (_Float16)0.0f;
  }
  __syncthreads();

  for (int s = 0; s < LL; ++s) {
    const int t = dir ? (LL - 1 - s) : s;

    // ---- gates = xs[:, t] (+bias already folded) + h @ Whh^T  ----
    v8f acc[2][4];
    #pragma unroll
    for (int mi = 0; mi < 2; ++mi)
      #pragma unroll
      for (int j = 0; j < 4; ++j) {
        const int col = nbase + j * 16 + lo;
        #pragma unroll
        for (int r = 0; r < 8; ++r) {
          const int b = mi * 16 + r + hi * 8;
          acc[mi][j][r] = xs[(size_t)(b * LL + t) * 2048 + dir * GATES + col];
        }
      }
    #pragma unroll
    for (int kk = 0; kk < HID_; kk += 32) {
      v16h af[2];
      #pragma unroll
      for (int mi = 0; mi < 2; ++mi)
        af[mi] = *(const v16h*)(hh + (mi * 16 + lo) * HID_ + kk + hi * 16);
      #pragma unroll
      for (int j = 0; j < 4; ++j) {
        const v16h b = *(const v16h*)(Whh + (size_t)(dir * GATES + nbase + j * 16 + lo) * HID_
                                          + kk + hi * 16);
        #pragma unroll
        for (int mi = 0; mi < 2; ++mi)
          acc[mi][j] = __builtin_amdgcn_wmma_f32_16x16x32_f16(false, af[mi], false, b,
                                                              (short)0, acc[mi][j],
                                                              false, false);
      }
    }
    #pragma unroll
    for (int mi = 0; mi < 2; ++mi)
      #pragma unroll
      for (int j = 0; j < 4; ++j) {
        const int col = nbase + j * 16 + lo;
        #pragma unroll
        for (int r = 0; r < 8; ++r)
          gt[(mi * 16 + r + hi * 8) * GATES + col] = acc[mi][j][r];
      }
    __syncthreads();

    // ---- elementwise cell update (i,f,g,o gate order as in reference) ----
    for (int e = tid; e < BB * HID_; e += 512) {
      const int b = e >> 8, j = e & 255;
      const float ig = 1.0f / (1.0f + expf(-gt[b * GATES + j]));
      const float fg = 1.0f / (1.0f + expf(-gt[b * GATES + 256 + j]));
      const float gg = tanhf(gt[b * GATES + 512 + j]);
      const float og = 1.0f / (1.0f + expf(-gt[b * GATES + 768 + j]));
      const float cn = fg * c[e] + ig * gg;
      const float hn = og * tanhf(cn);
      const float m  = mask[b * LL + t];
      const float hv = m * hn + (1.0f - m) * h[e];
      const float cv = m * cn + (1.0f - m) * c[e];
      h[e] = hv; c[e] = cv; hh[e] = (_Float16)hv;
      const size_t o = (size_t)(b * LL + t) * 512 + dir * HID_ + j;
      outF[o] = hv; outH[o] = (_Float16)hv;
    }
    __syncthreads();
  }
}

// ---------------------------------------------------------------------------
// CRF: one wave per (label k, batch b). Emissions em[t,n] computed as six
// 512-dots shuffle-reduced across the wave; lane 0 runs the 6-state forward
// recursion + gold score, and atomically accumulates (logZ - score)/B.
__global__ void k_crf(const float* __restrict__ out1,  // (B*L, 512)
                      const float* __restrict__ cw,    // (8, 512, 6)
                      const float* __restrict__ cbias, // (8, 6)
                      const float* __restrict__ ctr,   // (8, 6, 6)
                      const int* __restrict__ target,  // (8, B, L)
                      const float* __restrict__ mask,  // (B, L)
                      float* __restrict__ outLoss) {
  const int kb = blockIdx.x;        // 0..255
  const int k = kb >> 5, b = kb & 31;
  const int lane = threadIdx.x;     // 32 threads (wave32)
  float alpha[6];
  float score = 0.0f;
  int prevY = 0;

  for (int t = 0; t < LL; ++t) {
    float part[6] = {0, 0, 0, 0, 0, 0};
    const float* x = out1 + (size_t)(b * LL + t) * 512;
    for (int d = lane; d < 512; d += 32) {
      const float xv = x[d];
      const float* wrow = cw + (size_t)k * 3072 + d * 6;
      #pragma unroll
      for (int n = 0; n < 6; ++n) part[n] += xv * wrow[n];
    }
    #pragma unroll
    for (int n = 0; n < 6; ++n)
      for (int off = 16; off > 0; off >>= 1)
        part[n] += __shfl_down(part[n], off, 32);

    if (lane == 0) {
      float em[6];
      #pragma unroll
      for (int n = 0; n < 6; ++n) em[n] = part[n] + cbias[k * 6 + n];
      const int y = target[(size_t)(k * BB + b) * LL + t];
      const float m = mask[b * LL + t];
      score += em[y] * m;
      if (t > 0) score += ctr[k * 36 + prevY * 6 + y] * m;
      if (t == 0) {
        for (int n = 0; n < 6; ++n) alpha[n] = em[n];
      } else {
        float na[6];
        for (int j = 0; j < 6; ++j) {
          float mx = -1e30f;
          for (int i = 0; i < 6; ++i) mx = fmaxf(mx, alpha[i] + ctr[k * 36 + i * 6 + j]);
          float ssum = 0.0f;
          for (int i = 0; i < 6; ++i) ssum += expf(alpha[i] + ctr[k * 36 + i * 6 + j] - mx);
          na[j] = mx + logf(ssum) + em[j];
        }
        for (int j = 0; j < 6; ++j) alpha[j] = m * na[j] + (1.0f - m) * alpha[j];
      }
      prevY = y;
    }
  }
  if (lane == 0) {
    float mx = alpha[0];
    for (int i = 1; i < 6; ++i) mx = fmaxf(mx, alpha[i]);
    float ssum = 0.0f;
    for (int i = 0; i < 6; ++i) ssum += expf(alpha[i] - mx);
    const float logZ = mx + logf(ssum);
    atomicAdd(outLoss, (logZ - score) * (1.0f / (float)BB));
  }
}

// ---------------------------------------------------------------------------
extern "C" void kernel_launch(void* const* d_in, const int* in_sizes, int n_in,
                              void* d_out, int out_size, void* d_ws, size_t ws_size,
                              hipStream_t stream) {
  (void)in_sizes; (void)n_in; (void)out_size; (void)ws_size;

  const int*   iv     = (const int*)d_in[0];
  const int*   oo     = (const int*)d_in[1];
  const int*   chars  = (const int*)d_in[2];
  const int*   target = (const int*)d_in[3];
  const float* mask   = (const float*)d_in[4];
  const float* wtab   = (const float*)d_in[5];
  const float* otab   = (const float*)d_in[6];
  const float* ctab   = (const float*)d_in[7];
  const float* convw  = (const float*)d_in[8];
  const float* convb  = (const float*)d_in[9];
  const float* w_ih0  = (const float*)d_in[10];
  const float* w_hh0  = (const float*)d_in[11];
  const float* b0     = (const float*)d_in[12];
  const float* w_ih1  = (const float*)d_in[13];
  const float* w_hh1  = (const float*)d_in[14];
  const float* b1     = (const float*)d_in[15];
  const float* crf_w  = (const float*)d_in[16];
  const float* crf_b  = (const float*)d_in[17];
  const float* crf_tr = (const float*)d_in[18];

  // Workspace carve (all 256B aligned).
  char* ws = (char*)d_ws;
  size_t off = 0;
  auto carve = [&](size_t bytes) -> char* {
    char* p = ws + off;
    off += bytes;
    off = (off + 255) & ~(size_t)255;
    return p;
  };
  _Float16* Wih0  = (_Float16*)carve(2048ull * 512 * 2);
  _Float16* Wih1  = (_Float16*)carve(2048ull * 512 * 2);
  _Float16* Whh0  = (_Float16*)carve(2048ull * 256 * 2);
  _Float16* Whh1  = (_Float16*)carve(2048ull * 256 * 2);
  _Float16* inpH  = (_Float16*)carve(4096ull * 512 * 2);
  float*    xs    = (float*)   carve(4096ull * 2048 * 4);
  float*    out0F = (float*)   carve(4096ull * 512 * 4);
  _Float16* out0H = (_Float16*)carve(4096ull * 512 * 2);
  float*    out1F = (float*)   carve(4096ull * 512 * 4);
  _Float16* out1H = (_Float16*)carve(4096ull * 512 * 2);

  k_zero<<<1, 1, 0, stream>>>((float*)d_out);

  // Weights -> f16 (K padded to 512 for layer-0 input projection).
  k_cvt_pad<<<2048, 256, 0, stream>>>(w_ih0, Wih0, 500, 512);
  k_cvt_pad<<<2048, 256, 0, stream>>>(w_ih1, Wih1, 512, 512);
  k_cvt_pad<<<2048, 256, 0, stream>>>(w_hh0, Whh0, 256, 256);
  k_cvt_pad<<<2048, 256, 0, stream>>>(w_hh1, Whh1, 256, 256);

  // Features -> inp_f16 (4096 x 512): [0:300]=word, [300:500]=char CNN, [500:512]=0.
  k_embed<<<4096, 320, 0, stream>>>(iv, oo, wtab, otab, inpH);
  k_charconv<<<4096, 256, 0, stream>>>(chars, ctab, convw, convb, inpH);

  const dim3 gGemm(4096 / 32, 2048 / 256);
  const size_t ldsBytes = (size_t)(BB * HID_ * 2 + BB * GATES) * sizeof(float)
                        + (size_t)(BB * HID_) * sizeof(_Float16);  // 208 KB

  // Layer 0: xs = inp @ Wih0^T + b0 (both directions in N), then recurrence.
  k_wmma_gemm<<<gGemm, 256, 0, stream>>>(inpH, Wih0, b0, xs, 4096, 2048, 512);
  k_lstm<<<2, 512, ldsBytes, stream>>>(xs, Whh0, mask, out0F, out0H);

  // Layer 1 on concatenated [fwd,bwd] f16 output.
  k_wmma_gemm<<<gGemm, 256, 0, stream>>>(out0H, Wih1, b1, xs, 4096, 2048, 512);
  k_lstm<<<2, 512, ldsBytes, stream>>>(xs, Whh1, mask, out1F, out1H);

  // CRF losses, accumulated into the scalar output.
  k_crf<<<256, 32, 0, stream>>>(out1F, crf_w, crf_b, crf_tr, target, mask,
                                (float*)d_out);
}
